// Inv2d_13503377178771
// MI455X (gfx1250) — matrix operand order repaired and checked by reading, not compiled, not run on hardware
//
#include <hip/hip_runtime.h>
#include <hip/hip_bf16.h>
#include <stddef.h>

// ---------------- CDNA5 WMMA plumbing (gfx1250, wave32) ----------------
typedef __attribute__((ext_vector_type(16))) __bf16        v16bf;
typedef __attribute__((ext_vector_type(8)))  float         v8f;
typedef __attribute__((ext_vector_type(8)))  unsigned int  v8u;
typedef __attribute__((ext_vector_type(4)))  unsigned int  v4u;

static __device__ __forceinline__ v8f wmma_bf16(v8u a, v8u b, v8f c) {
  // D = A(16x32 bf16) * B(32x16 bf16) + C(16x16 f32)
  return __builtin_amdgcn_wmma_f32_16x16x32_bf16(
      false, __builtin_bit_cast(v16bf, a),
      false, __builtin_bit_cast(v16bf, b),
      (short)0, c, false, false);
}

static __device__ __forceinline__ unsigned short f32_bf16(float f) {
  unsigned int u = __builtin_bit_cast(unsigned int, f);
  unsigned int r = u + 0x7FFFu + ((u >> 16) & 1u);   // round-to-nearest-even
  return (unsigned short)(r >> 16);
}

// Problem constants
#define BB   8
#define CH   256
#define KK2  9       // 3x3
#define KCH  144     // K*K*G rows of W_span / ker channels
#define PADW 66      // 64 + 2
#define XTE  4368    // 66*66=4356 padded to 16B multiple (shorts)
#define CSTR 36      // LDS copy column stride in dwords (72 shorts = 144B, 16B-aligned)
#define CPYD (PADW * CSTR)  // 2376 dwords per shifted copy

// workspace byte offsets
#define OFF_WCOMB 0u                 // 144*256 bf16      = 73728 B
#define OFF_BCOMB 81920u             // 144 f32           = 576 B
#define OFF_KER   131072u            // 8*144*4096 bf16   = 9437184 B
#define OFF_XPT   9568256u           // 2048*4368 bf16    = 17891328 B
#define OFF_XCL   27459584u          // 8*4096*256 bf16   = 16777216 B (channel-last x)

// ---------------- kernel 0: fold the two 1x1 convs into one ----------------
__global__ __launch_bounds__(256) void combine_kernel(
    const float* __restrict__ Wr, const float* __restrict__ br,
    const float* __restrict__ Wsp, const float* __restrict__ bsp,
    unsigned short* __restrict__ wcomb, float* __restrict__ bcomb) {
  const int kk = blockIdx.x;        // 0..143
  const int c  = threadIdx.x;       // 0..255
  float acc = 0.f;
  #pragma unroll 8
  for (int o = 0; o < 128; ++o)
    acc += Wsp[kk * 128 + o] * Wr[o * 256 + c];
  wcomb[kk * 256 + c] = f32_bf16(acc);
  if (c == 0) {
    float bb = bsp[kk];
    for (int o = 0; o < 128; ++o) bb += Wsp[kk * 128 + o] * br[o];
    bcomb[kk] = bb;
  }
}

// ------- kernel 1: per-channel zero-padded TRANSPOSED bf16 image xT[u][w] -------
// xT[u][w] = x[w-1][u-1] (0 on the 1-wide border). Transposed so the WMMA
// B-fragment K-pairs (rows of the shifted image) are contiguous 16-bit pairs.
__global__ __launch_bounds__(256) void pad_kernel(
    const float* __restrict__ x, unsigned short* __restrict__ xpt) {
  const int bc = blockIdx.x;                    // b*256+ch
  const float* xs = x + ((size_t)bc << 12);
  unsigned short* dst = xpt + (size_t)bc * XTE;
  for (int e = threadIdx.x; e < XTE; e += 256) {
    const int u = e / PADW, w = e - u * PADW;
    float val = 0.f;
    if (e < PADW * PADW && u >= 1 && u <= 64 && w >= 1 && w <= 64)
      val = xs[(w - 1) * 64 + (u - 1)];
    dst[e] = f32_bf16(val);
  }
}

// ------- kernel 2: channel-last bf16 copy: xcl[b][hw][c] = bf16(x[b][c][hw]) -------
// Makes gemmA's WMMA B-fragments contiguous (8 dwords -> 2 x global_load_b128).
__global__ __launch_bounds__(256) void xpose_kernel(
    const float* __restrict__ x, unsigned short* __restrict__ xcl) {
  __shared__ unsigned short tile[64 * 65];      // [cL][hwL], stride 65 vs conflicts
  const int hw0 = blockIdx.x << 6;              // 64 hw per tile
  const int c0  = blockIdx.y << 6;              // 64 c  per tile
  const int b   = blockIdx.z;
  const int t   = threadIdx.x;
  const int hwL = t & 63, tq = t >> 6;          // tq in 0..3
  #pragma unroll
  for (int r = 0; r < 16; ++r) {                // coalesced f32 reads along hw
    const int cL = (r << 2) + tq;
    tile[cL * 65 + hwL] =
        f32_bf16(x[((size_t)((b << 8) + c0 + cL) << 12) + hw0 + hwL]);
  }
  __syncthreads();
  const int cL2 = t & 63;
  #pragma unroll
  for (int r = 0; r < 16; ++r) {                // coalesced bf16 writes along c
    const int hwL2 = (r << 2) + tq;
    xcl[((size_t)((b << 12) + hw0 + hwL2) << 8) + c0 + cL2] =
        tile[cL2 * 65 + hwL2];
  }
}

// ------- kernel 3: ker = W_comb[144x256] @ x[256 x 32768] + b_comb  (bf16 out) -------
__global__ __launch_bounds__(256) void gemmA_kernel(
    const unsigned int* __restrict__ xcl32, const unsigned int* __restrict__ wc32,
    const float* __restrict__ bcomb, unsigned short* __restrict__ ker) {
  const int m0    = blockIdx.x << 4;            // 0..128 (9 M-tiles)
  const int wave  = threadIdx.x >> 5;
  const int lane  = threadIdx.x & 31;
  const int ntile = (blockIdx.y << 3) + wave;   // 0..2047
  const int n0    = ntile << 4;
  const int bidx  = n0 >> 12;                   // batch (tile never crosses b)
  const int hw    = (n0 & 4095) + (lane & 15);
  const int hi    = lane >> 4;                  // K-half select
  const int khalf = hi << 3;                    // unused; A also uses kb (+16)
  const int kb    = hi << 4;                    // B: +16
  const int kkrow = m0 + (lane & 15);           // A row for this lane
  const int kv[8] = {0, 2, 4, 6, 16, 18, 20, 22};
  const size_t rowb = ((size_t)(bidx << 12) + hw) << 7;  // dword row base in xcl

  v8f acc = {};
  #pragma unroll
  for (int kt = 0; kt < 8; ++kt) {              // K = 256 in chunks of 32
    const int c0 = kt << 5;
    v8u au, bu;
    #pragma unroll
    for (int v = 0; v < 8; ++v)                 // A frag: 2 x b128 (contiguous pairs)
      au[v] = wc32[kkrow * 128 + ((c0 + kb) >> 1) + v];
    #pragma unroll
    for (int v = 0; v < 8; ++v)                 // B frag: 2 x b128 from channel-last x
      bu[v] = xcl32[rowb + ((c0 + kb) >> 1) + v];
    acc = wmma_bf16(au, bu, acc);
  }
  #pragma unroll
  for (int r = 0; r < 8; ++r) {                 // D: lanes16-31 are M+8
    const int kkr = m0 + r + (hi << 3);
    ker[((size_t)(bidx * KCH + kkr) << 12) + hw] = f32_bf16(acc[r] + bcomb[kkr]);
  }
}

// ------- kernel 4: Out[b,ch] = sum_k Ker[b,g,k](64x64) @ Shift_k(x[b,ch]) -------
// LDS holds THREE row-shift copies of the transposed padded image (one per ki),
// column stride 72 shorts, so every WMMA B-fragment is 2 aligned ds_load_b128.
__global__ __launch_bounds__(256) void invB_kernel(
    const unsigned int* __restrict__ xpt32, const unsigned int* __restrict__ ker32,
    float* __restrict__ out) {
  __shared__ __align__(16) unsigned int xs[3 * CPYD];   // 28512 B
  const int ch = blockIdx.x;
  const int b  = blockIdx.y;
  const int g  = ch >> 4;

  // Build shifted copies: xs[ki][u][t(dword)] packs xp[2t+ki .. 2t+1+ki][u].
  // Source column is 33 dwords (66 shorts); padded tail of xpt is zero.
  const unsigned int* src = xpt32 + (size_t)(b * 256 + ch) * (XTE / 2);
  for (int idx = threadIdx.x; idx < 3 * CPYD; idx += 256) {
    const int ki  = idx / CPYD;
    const int rem = idx - ki * CPYD;
    const int u   = rem / CSTR, d = rem - u * CSTR;
    unsigned int w = 0;
    if (d < 33) {                               // only dwords 0..32 ever read
      const int s = u * 33 + d;
      if (ki == 0)      w = src[s];
      else if (ki == 2) w = src[s + 1];
      else              w = (src[s] >> 16) | (src[s + 1] << 16);
    }
    xs[idx] = w;
  }
  __builtin_prefetch(ker32 + (((size_t)(b * KCH + g * KK2) << 12) >> 1), 0, 1);
  __syncthreads();

  const int wave  = threadIdx.x >> 5;
  const int lane  = threadIdx.x & 31;
  const int i0    = (wave >> 1) << 4;           // output tile row
  const int j0    = (wave & 1) << 5;            // two tiles: j0, j0+16
  const int hi    = lane >> 4;
  const int khalf = hi << 3;                    // unused; A also uses kb (+16)
  const int kb    = hi << 4;                    // B K-half: +16 (shorts)
  const int nlo   = lane & 15;
  const int arow  = (i0 + (lane & 15)) << 6;    // A row offset (i * 64)
  const int kv[8] = {0, 2, 4, 6, 16, 18, 20, 22};

  v8f acc0 = {}, acc1 = {};
  #pragma unroll
  for (int k = 0; k < KK2; ++k) {               // 9 shifts, fully unrolled
    const int ki = k / 3, kj = k - 3 * (k / 3);
    const size_t chanA = (size_t)(b * KCH + g * KK2 + k) << 12;
    const unsigned int* bsrc = xs + ki * CPYD;
    const int u0d = (j0 + nlo + kj) * CSTR;         // tile 0 column base (dwords)
    const int u1d = u0d + 16 * CSTR;                // tile 1
    #pragma unroll
    for (int m0c = 0; m0c < 64; m0c += 32) {    // K = 64 per shift
      v8u au, bu0, bu1;
      #pragma unroll
      for (int v = 0; v < 8; ++v)               // A = Ker tile: 2 x global b128
        au[v] = ker32[(chanA + arow + m0c + kb + 2 * v) >> 1];
      const int boff = (m0c + kb) >> 1;         // dwords; keeps 16B alignment
      const v4u* p0 = (const v4u*)(bsrc + u0d + boff);
      const v4u* p1 = (const v4u*)(bsrc + u1d + boff);
      const v4u lo0 = p0[0], hi0 = p0[1], lo1 = p1[0], hi1 = p1[1];
      #pragma unroll
      for (int v = 0; v < 4; ++v) {
        bu0[v] = lo0[v]; bu0[v + 4] = hi0[v];
        bu1[v] = lo1[v]; bu1[v + 4] = hi1[v];
      }
      acc0 = wmma_bf16(au, bu0, acc0);
      acc1 = wmma_bf16(au, bu1, acc1);
    }
  }
  float* op = out + ((size_t)(b * 256 + ch) << 12);
  #pragma unroll
  for (int r = 0; r < 8; ++r) {
    const int i = i0 + r + (hi << 3);
    op[i * 64 + j0 + nlo]      = acc0[r];
    op[i * 64 + j0 + 16 + nlo] = acc1[r];
  }
}

// ---------------------------- launcher ----------------------------
extern "C" void kernel_launch(void* const* d_in, const int* in_sizes, int n_in,
                              void* d_out, int out_size, void* d_ws, size_t ws_size,
                              hipStream_t stream) {
  (void)in_sizes; (void)n_in; (void)out_size; (void)ws_size;
  const float* x   = (const float*)d_in[0];
  const float* Wr  = (const float*)d_in[1];
  const float* br  = (const float*)d_in[2];
  const float* Wsp = (const float*)d_in[3];
  const float* bsp = (const float*)d_in[4];
  float* out = (float*)d_out;

  unsigned char* ws = (unsigned char*)d_ws;
  unsigned short* wcomb = (unsigned short*)(ws + OFF_WCOMB);
  float*          bcomb = (float*)(ws + OFF_BCOMB);
  unsigned short* ker   = (unsigned short*)(ws + OFF_KER);
  unsigned short* xpt   = (unsigned short*)(ws + OFF_XPT);
  unsigned short* xcl   = (unsigned short*)(ws + OFF_XCL);

  combine_kernel<<<dim3(KCH), dim3(256), 0, stream>>>(Wr, br, Wsp, bsp, wcomb, bcomb);
  pad_kernel<<<dim3(BB * CH), dim3(256), 0, stream>>>(x, xpt);
  xpose_kernel<<<dim3(64, 4, BB), dim3(256), 0, stream>>>(x, xcl);
  gemmA_kernel<<<dim3(KCH / 16, 256), dim3(256), 0, stream>>>(
      (const unsigned int*)xcl, (const unsigned int*)wcomb, bcomb, ker);
  invB_kernel<<<dim3(CH, BB), dim3(256), 0, stream>>>(
      (const unsigned int*)xpt, (const unsigned int*)ker, out);
}
